// GraphSAGE_DDI_64622077935661
// MI455X (gfx1250) — matrix operand order repaired and checked by reading, compiled, not run on hardware
//
#include <hip/hip_runtime.h>

typedef __bf16 v16bf __attribute__((ext_vector_type(16)));
typedef float  v8f   __attribute__((ext_vector_type(8)));

#define DD 128            // node/hidden dim
#define KK2 256           // GNN GEMM K = 2*DD
#define KMLP 384          // MLP K = 2*DD + kge_dim
#define LAYERS 3

// hardware f32 -> bf16 (lowers to v_cvt_pk_bf16_f32 pairs)
__device__ __forceinline__ __bf16 tobf(float f) { return (__bf16)f; }

__device__ __forceinline__ v16bf pack_a_scaled(float4 c0, float4 c1, float4 c2,
                                               float4 c3, float am) {
    v16bf a;
    a[0]  = tobf(c0.x * am); a[1]  = tobf(c0.y * am);
    a[2]  = tobf(c0.z * am); a[3]  = tobf(c0.w * am);
    a[4]  = tobf(c1.x * am); a[5]  = tobf(c1.y * am);
    a[6]  = tobf(c1.z * am); a[7]  = tobf(c1.w * am);
    a[8]  = tobf(c2.x * am); a[9]  = tobf(c2.y * am);
    a[10] = tobf(c2.z * am); a[11] = tobf(c2.w * am);
    a[12] = tobf(c3.x * am); a[13] = tobf(c3.y * am);
    a[14] = tobf(c3.z * am); a[15] = tobf(c3.w * am);
    return a;
}

__device__ __forceinline__ v16bf pack_a_plain(float4 c0, float4 c1, float4 c2,
                                              float4 c3) {
    v16bf a;
    a[0]  = tobf(c0.x); a[1]  = tobf(c0.y);
    a[2]  = tobf(c0.z); a[3]  = tobf(c0.w);
    a[4]  = tobf(c1.x); a[5]  = tobf(c1.y);
    a[6]  = tobf(c1.z); a[7]  = tobf(c1.w);
    a[8]  = tobf(c2.x); a[9]  = tobf(c2.y);
    a[10] = tobf(c2.z); a[11] = tobf(c2.w);
    a[12] = tobf(c3.x); a[13] = tobf(c3.y);
    a[14] = tobf(c3.z); a[15] = tobf(c3.w);
    return a;
}

// ---------------- prep: transpose+convert weights to bf16, fold BN ----------
__global__ void prep_kernel(const float* __restrict__ Wl, const float* __restrict__ Wr,
                            const float* __restrict__ W1, const float* __restrict__ blv,
                            const float* __restrict__ gam, const float* __restrict__ bet,
                            const float* __restrict__ mean, const float* __restrict__ var,
                            __bf16* __restrict__ wt,      // [L][128][256]
                            __bf16* __restrict__ w1t,     // [128][384]
                            float* __restrict__ scale, float* __restrict__ shift) {
    int tid = blockIdx.x * blockDim.x + threadIdx.x;
    int stride = gridDim.x * blockDim.x;
    for (int i = tid; i < LAYERS * DD * KK2; i += stride) {
        int l = i / (DD * KK2);
        int rem = i - l * (DD * KK2);
        int n = rem / KK2;
        int k = rem - n * KK2;
        float v = (k < DD) ? Wl[((size_t)l * DD + k) * DD + n]
                           : Wr[((size_t)l * DD + (k - DD)) * DD + n];
        wt[i] = tobf(v);
    }
    for (int i = tid; i < DD * KMLP; i += stride) {
        int n = i / KMLP;
        int k = i - n * KMLP;
        w1t[i] = tobf(W1[(size_t)k * DD + n]);
    }
    for (int i = tid; i < LAYERS * DD; i += stride) {
        float s = gam[i] * rsqrtf(var[i] + 1e-5f);
        scale[i] = s;
        shift[i] = (blv[i] - mean[i]) * s + bet[i];
    }
}

// ---------------- degree ----------------------------------------------------
__global__ void deg_kernel(const int* __restrict__ ei1, const int* __restrict__ ei2,
                           float* __restrict__ deg1, float* __restrict__ deg2, int E) {
    int tid = blockIdx.x * blockDim.x + threadIdx.x;
    if (tid >= 2 * E) return;
    int g = tid / E;
    int e = tid - g * E;
    const int* ei = g ? ei2 : ei1;
    float* deg = g ? deg2 : deg1;
    atomicAdd(&deg[ei[E + e]], 1.0f);
}

// ---------------- rdeg: deg <- 1/max(deg,1) (fast rcp, bf16-accurate) -------
__global__ void rdeg_kernel(float* __restrict__ deg1, float* __restrict__ deg2, int N) {
    int tid = blockIdx.x * blockDim.x + threadIdx.x;
    if (tid >= 2 * N) return;
    float* d = (tid < N) ? deg1 : deg2;
    int i = (tid < N) ? tid : (tid - N);
    float v = d[i];
    d[i] = __builtin_amdgcn_rcpf(fmaxf(v, 1.0f));
}

// ---------------- edge scatter: agg[dst] += x[src] --------------------------
__global__ void scatter_kernel(const int* __restrict__ ei1, const int* __restrict__ ei2,
                               const float* __restrict__ x1, const float* __restrict__ x2,
                               float* __restrict__ agg1, float* __restrict__ agg2, int E) {
    long tid = (long)blockIdx.x * blockDim.x + threadIdx.x;
    long per = (long)E * 32;
    if (tid >= 2 * per) return;
    int g = (int)(tid / per);
    long r = tid - (long)g * per;
    int e = (int)(r >> 5);
    int c = (int)(r & 31);
    const int* ei = g ? ei2 : ei1;
    const float* x = g ? x2 : x1;
    float* agg = g ? agg2 : agg1;
    int src = ei[e];
    int dst = ei[E + e];
    const float4 v = *(const float4*)(x + (size_t)src * DD + c * 4);
    float* ap = agg + (size_t)dst * DD + c * 4;
    atomicAdd(ap + 0, v.x);
    atomicAdd(ap + 1, v.y);
    atomicAdd(ap + 2, v.z);
    atomicAdd(ap + 3, v.w);
}

// ---------------- WMMA GEMM: relu(BN([agg*rdeg | x] @ [Wl;Wr] + bl)) --------
// One wave computes a 32x128 output strip (two 16-row sub-tiles sharing every
// B fragment -> 2 WMMAs per weight fetch, 16 f32 accumulators).
__global__ __launch_bounds__(256) void gemm_bn_relu_kernel(
    const float* __restrict__ agg1, const float* __restrict__ agg2,
    const float* __restrict__ xin1, const float* __restrict__ xin2,
    float* __restrict__ out1, float* __restrict__ out2,
    const float* __restrict__ rdeg1, const float* __restrict__ rdeg2,
    const __bf16* __restrict__ wt,   // [128][256] bf16 (layer slice)
    const float* __restrict__ scale, const float* __restrict__ shift,
    int ntiles32, int N) {
    const int lane = threadIdx.x & 31;
    const int wave = threadIdx.x >> 5;
    const int tile = blockIdx.x * 8 + wave;
    if (tile >= ntiles32) return;
    const int g = blockIdx.y;
    const float* agg   = g ? agg2 : agg1;
    const float* xin   = g ? xin2 : xin1;
    float* out         = g ? out2 : out1;
    const float* rdeg  = g ? rdeg2 : rdeg1;

    const int m0 = tile * 32;
    const int rowA = m0 + (lane & 15);
    int rowB = m0 + 16 + (lane & 15);
    if (rowB > N - 1) rowB = N - 1;           // clamp (keeps EXEC full for WMMA)
    const int half = lane >> 4;
    const int kb = half * 8;
    const float rdA = rdeg[rowA];
    const float rdB = rdeg[rowB];

    v8f zero = {0.f, 0.f, 0.f, 0.f, 0.f, 0.f, 0.f, 0.f};
    v8f acc[16];
#pragma unroll
    for (int nt = 0; nt < 16; ++nt) acc[nt] = zero;

#pragma unroll
    for (int k0 = 0; k0 < KK2; k0 += 32) {
        const bool isAgg = (k0 < DD);
        const float* base = isAgg ? agg : xin;
        const int kcol = isAgg ? k0 : (k0 - DD);
        const float* aA = base + (size_t)rowA * DD + kcol;
        const float* aB = base + (size_t)rowB * DD + kcol;
        float4 a0 = *(const float4*)(aA + 2 * kb);
        float4 a1 = *(const float4*)(aA + 2 * kb + 4);
        float4 a2 = *(const float4*)(aA + 2 * kb + 8);
        float4 a3 = *(const float4*)(aA + 2 * kb + 12);
        float4 b0 = *(const float4*)(aB + 2 * kb);
        float4 b1 = *(const float4*)(aB + 2 * kb + 4);
        float4 b2 = *(const float4*)(aB + 2 * kb + 8);
        float4 b3 = *(const float4*)(aB + 2 * kb + 12);
        v16bf fragA = isAgg ? pack_a_scaled(a0, a1, a2, a3, rdA)
                            : pack_a_plain(a0, a1, a2, a3);
        v16bf fragB = isAgg ? pack_a_scaled(b0, b1, b2, b3, rdB)
                            : pack_a_plain(b0, b1, b2, b3);
        const __bf16* brow = wt + k0 + half * 16;
#pragma unroll
        for (int nt = 0; nt < 8; ++nt) {
            v16bf bm = *(const v16bf*)(const void*)(brow + (size_t)(nt * 16 + (lane & 15)) * KK2);
            acc[nt] = __builtin_amdgcn_wmma_f32_16x16x32_bf16(
                false, fragA, false, bm, (short)0, acc[nt], false, false);
            acc[8 + nt] = __builtin_amdgcn_wmma_f32_16x16x32_bf16(
                false, fragB, false, bm, (short)0, acc[8 + nt], false, false);
        }
    }
#pragma unroll
    for (int nt = 0; nt < 8; ++nt) {
        int n = nt * 16 + (lane & 15);
        float s = scale[n], t = shift[n];
#pragma unroll
        for (int r = 0; r < 8; ++r) {
            int m = m0 + r + half * 8;
            float y = acc[nt][r] * s + t;
            out[(size_t)m * DD + n] = fmaxf(y, 0.0f);
        }
#pragma unroll
        for (int r = 0; r < 8; ++r) {
            int m = m0 + 16 + r + half * 8;
            float y = acc[8 + nt][r] * s + t;
            if (m < N) out[(size_t)m * DD + n] = fmaxf(y, 0.0f);
        }
    }
}

// ---------------- global add pool -------------------------------------------
__global__ void pool_kernel(const float* __restrict__ xb1, const float* __restrict__ xb2,
                            const int* __restrict__ b1i, const int* __restrict__ b2i,
                            float* __restrict__ p1, float* __restrict__ p2, int N) {
    long tid = (long)blockIdx.x * blockDim.x + threadIdx.x;
    long per = (long)N * 32;
    if (tid >= 2 * per) return;
    int g = (int)(tid / per);
    long r = tid - (long)g * per;
    int n = (int)(r >> 5);
    int c = (int)(r & 31);
    const float* x = g ? xb2 : xb1;
    const int* bi = g ? b2i : b1i;
    float* p = g ? p2 : p1;
    int seg = bi[n];
    const float4 v = *(const float4*)(x + (size_t)n * DD + c * 4);
    float* pp = p + (size_t)seg * DD + c * 4;
    atomicAdd(pp + 0, v.x);
    atomicAdd(pp + 1, v.y);
    atomicAdd(pp + 2, v.z);
    atomicAdd(pp + 3, v.w);
}

// ---------------- final MLP (WMMA stage1 + LDS dot stage2) ------------------
__global__ __launch_bounds__(128) void mlp_kernel(
    const float* __restrict__ p1, const float* __restrict__ p2,
    const float* __restrict__ kge, const int* __restrict__ rel,
    const __bf16* __restrict__ w1t,            // [128][384] bf16
    const float* __restrict__ b1v, const float* __restrict__ w2,
    const float* __restrict__ b2v, float* __restrict__ outv, int ntiles) {
    __shared__ float hl[4][16][DD];
    const int lane = threadIdx.x & 31;
    const int wave = threadIdx.x >> 5;
    const int tile = blockIdx.x * 4 + wave;
    if (tile >= ntiles) return;
    const int m0 = tile * 16;
    const int mrow = m0 + (lane & 15);
    const int half = lane >> 4;
    const int kb = half * 8;
    const int rr = rel[mrow];

    v8f zero = {0.f, 0.f, 0.f, 0.f, 0.f, 0.f, 0.f, 0.f};
    v8f acc[8];
#pragma unroll
    for (int nt = 0; nt < 8; ++nt) acc[nt] = zero;

#pragma unroll
    for (int k0 = 0; k0 < KMLP; k0 += 32) {
        const float* abuf;
        if (k0 < DD)            abuf = p1  + (size_t)mrow * DD + k0;
        else if (k0 < 2 * DD)   abuf = p2  + (size_t)mrow * DD + (k0 - DD);
        else                    abuf = kge + (size_t)rr   * DD + (k0 - 2 * DD);
        float4 c0 = *(const float4*)(abuf + 2 * kb);
        float4 c1 = *(const float4*)(abuf + 2 * kb + 4);
        float4 c2 = *(const float4*)(abuf + 2 * kb + 8);
        float4 c3 = *(const float4*)(abuf + 2 * kb + 12);
        v16bf fragA = pack_a_plain(c0, c1, c2, c3);
        const __bf16* brow = w1t + k0 + half * 16;
#pragma unroll
        for (int nt = 0; nt < 8; ++nt) {
            v16bf bm = *(const v16bf*)(const void*)(brow + (size_t)(nt * 16 + (lane & 15)) * KMLP);
            acc[nt] = __builtin_amdgcn_wmma_f32_16x16x32_bf16(
                false, fragA, false, bm, (short)0, acc[nt], false, false);
        }
    }
    // epilogue: relu(acc + b1) -> LDS
#pragma unroll
    for (int nt = 0; nt < 8; ++nt) {
        int n = nt * 16 + (lane & 15);
        float bb = b1v[n];
#pragma unroll
        for (int r = 0; r < 8; ++r) {
            int m = r + half * 8;
            hl[wave][m][n] = fmaxf(acc[nt][r] + bb, 0.0f);
        }
    }
    __syncthreads();
    // stage 2: out[b] = h[b] . W2 + b2   (2 lanes per row, 64 cols each)
    int r2 = lane >> 1;
    int ch = lane & 1;
    float part = 0.0f;
#pragma unroll 8
    for (int j = 0; j < 64; ++j) part += hl[wave][r2][ch * 64 + j] * w2[ch * 64 + j];
    part += __shfl_xor(part, 1, 32);
    if (ch == 0) outv[m0 + r2] = part + b2v[0];
}

// ---------------- launch ----------------------------------------------------
#define RND512(x) ((((size_t)(x)) + 511) & ~(size_t)511)

extern "C" void kernel_launch(void* const* d_in, const int* in_sizes, int n_in,
                              void* d_out, int out_size, void* d_ws, size_t ws_size,
                              hipStream_t stream) {
    const float* x1  = (const float*)d_in[0];
    const float* x2  = (const float*)d_in[1];
    const int* ei1   = (const int*)d_in[2];
    const int* ei2   = (const int*)d_in[3];
    const int* b1i   = (const int*)d_in[4];
    const int* b2i   = (const int*)d_in[5];
    const int* rel   = (const int*)d_in[6];
    const float* Wl  = (const float*)d_in[7];
    const float* blv = (const float*)d_in[8];
    const float* Wr  = (const float*)d_in[9];
    const float* gam = (const float*)d_in[10];
    const float* bet = (const float*)d_in[11];
    const float* mea = (const float*)d_in[12];
    const float* var = (const float*)d_in[13];
    const float* kge = (const float*)d_in[14];
    const float* W1  = (const float*)d_in[15];
    const float* b1v = (const float*)d_in[16];
    const float* W2  = (const float*)d_in[17];
    const float* b2v = (const float*)d_in[18];
    float* out = (float*)d_out;

    const int N = in_sizes[4];        // 50000
    const int E = in_sizes[2] / 2;    // 600000
    const int B = in_sizes[6];        // 4096

    char* ws = (char*)d_ws;
    size_t o = 0;
    __bf16* wt  = (__bf16*)(ws + o); o += RND512((size_t)LAYERS * DD * KK2 * 2);
    __bf16* w1t = (__bf16*)(ws + o); o += RND512((size_t)DD * KMLP * 2);
    float* scale = (float*)(ws + o); o += RND512((size_t)LAYERS * DD * 4);
    float* shift = (float*)(ws + o); o += RND512((size_t)LAYERS * DD * 4);
    float* deg1  = (float*)(ws + o); o += RND512((size_t)N * 4);
    float* deg2  = (float*)(ws + o); o += RND512((size_t)N * 4);
    float* agg1  = (float*)(ws + o); o += RND512((size_t)N * DD * 4);
    float* agg2  = (float*)(ws + o); o += RND512((size_t)N * DD * 4);
    float* xb1   = (float*)(ws + o); o += RND512((size_t)N * DD * 4);
    float* xb2   = (float*)(ws + o); o += RND512((size_t)N * DD * 4);
    float* p1    = (float*)(ws + o); o += RND512((size_t)B * DD * 4);
    float* p2    = (float*)(ws + o); o += RND512((size_t)B * DD * 4);
    (void)ws_size; (void)n_in; (void)out_size;

    prep_kernel<<<192, 256, 0, stream>>>(Wl, Wr, W1, blv, gam, bet, mea, var,
                                         wt, w1t, scale, shift);

    hipMemsetAsync(deg1, 0, (size_t)N * 4, stream);
    hipMemsetAsync(deg2, 0, (size_t)N * 4, stream);
    deg_kernel<<<(2 * E + 255) / 256, 256, 0, stream>>>(ei1, ei2, deg1, deg2, E);
    rdeg_kernel<<<(2 * N + 255) / 256, 256, 0, stream>>>(deg1, deg2, N);

    const int ntiles16 = N / 16;                    // 3125 (N divisible by 16)
    const int ntiles32 = (ntiles16 + 1) / 2;        // 1563
    const long sthreads = 2L * E * 32;
    for (int l = 0; l < LAYERS; ++l) {
        hipMemsetAsync(agg1, 0, (size_t)N * DD * 4, stream);
        hipMemsetAsync(agg2, 0, (size_t)N * DD * 4, stream);
        const float* xin1 = (l == 0) ? x1 : xb1;
        const float* xin2 = (l == 0) ? x2 : xb2;
        scatter_kernel<<<(unsigned)((sthreads + 255) / 256), 256, 0, stream>>>(
            ei1, ei2, xin1, xin2, agg1, agg2, E);
        dim3 gg((ntiles32 + 7) / 8, 2);
        gemm_bn_relu_kernel<<<gg, 256, 0, stream>>>(
            agg1, agg2, xin1, xin2, xb1, xb2, deg1, deg2,
            wt + (size_t)l * DD * KK2, scale + l * DD, shift + l * DD, ntiles32, N);
    }

    hipMemsetAsync(p1, 0, (size_t)B * DD * 4, stream);
    hipMemsetAsync(p2, 0, (size_t)B * DD * 4, stream);
    const long pthreads = 2L * N * 32;
    pool_kernel<<<(unsigned)((pthreads + 255) / 256), 256, 0, stream>>>(
        xb1, xb2, b1i, b2i, p1, p2, N);

    const int btiles = B / 16;                      // 256
    mlp_kernel<<<(btiles + 3) / 4, 128, 0, stream>>>(
        p1, p2, kge, rel, w1t, b1v, W2, b2v, out, btiles);
}
